// RSUT5_61701500174699
// MI455X (gfx1250) — matrix operand order repaired and checked by reading, not compiled, not run on hardware
//
#include <hip/hip_runtime.h>
#include <hip/hip_bf16.h>
#include <math.h>

typedef _Float16 h16;
typedef __attribute__((ext_vector_type(16))) _Float16 v16h;
typedef __attribute__((ext_vector_type(8)))  _Float16 v8h;
typedef __attribute__((ext_vector_type(8)))  float    v8f;
typedef __attribute__((ext_vector_type(4)))  float    v4f;
typedef __attribute__((ext_vector_type(4)))  int      v4i;

#define LTOK 32768
#define C0   256
#define BM 128
#define BN 128
#define BK 32

// ---- CDNA5 async global->LDS path (ASYNCcnt domain), with safe fallback ----
#if defined(__HIP_DEVICE_COMPILE__) && __has_builtin(__builtin_amdgcn_global_load_async_to_lds_b128)
#define HAVE_ASYNC_LDS 1
typedef __attribute__((address_space(1))) v4i* gp4_t;     // non-const, per builtin signature
typedef __attribute__((address_space(3))) v4i* lp4_t;
#define ASYNC_CP16(ldst, gsrc) \
    __builtin_amdgcn_global_load_async_to_lds_b128((gp4_t)(gsrc), (lp4_t)(ldst), 0, 0)
#if __has_builtin(__builtin_amdgcn_s_wait_asynccnt)
#define WAIT_ASYNC(n) __builtin_amdgcn_s_wait_asynccnt(n)
#else
#define WAIT_ASYNC(n) asm volatile("s_wait_asynccnt " #n ::: "memory")
#endif
#else
#define HAVE_ASYNC_LDS 0
#define ASYNC_CP16(ldst, gsrc) (*(v8h*)(ldst) = *(const v8h*)(gsrc))
#define WAIT_ASYNC(n) ((void)0)
#endif

// ---------------- fp32 -> fp16 weight conversion ----------------
__global__ void cvt_f16_kernel(const float* __restrict__ src, h16* __restrict__ dst, int n) {
    int i = blockIdx.x * blockDim.x + threadIdx.x;
    if (i < n) dst[i] = (h16)src[i];
}

// ---------------- LayerNorm (+optional shift+window partition), fp32 -> fp16 ----------------
// one wave per output row; blockDim = 256 (8 waves)
__global__ void ln_part_kernel(const float* __restrict__ x, const float* __restrict__ g,
                               const float* __restrict__ b, h16* __restrict__ out,
                               int shift, int windowed) {
    int wv   = threadIdx.x >> 5;
    int lane = threadIdx.x & 31;
    int orow = blockIdx.x * 8 + wv;
    int src;
    if (windowed) {
        int win = orow >> 6, n = orow & 63;
        int wh = win >> 6, ww = (win >> 3) & 7, wt = win & 7;
        int i = n >> 4, j = (n >> 2) & 3, k = n & 3;
        int h = ((wh << 2) + i + shift) & 31;
        int w = ((ww << 2) + j + shift) & 31;
        int t = ((wt << 2) + k + shift) & 31;
        src = ((h << 5) + w) * 32 + t;
    } else {
        src = orow;
    }
    const float* xr = x + (size_t)src * C0;
    int cb = lane * 8;
    v4f a0 = *(const v4f*)(xr + cb);
    v4f a1 = *(const v4f*)(xr + cb + 4);
    float s = 0.f, ss = 0.f;
    for (int i = 0; i < 4; i++) { s += a0[i] + a1[i]; ss += a0[i]*a0[i] + a1[i]*a1[i]; }
    for (int off = 1; off < 32; off <<= 1) { s += __shfl_xor(s, off, 32); ss += __shfl_xor(ss, off, 32); }
    float mean = s * (1.0f / C0);
    float var  = ss * (1.0f / C0) - mean * mean;
    float rstd = rsqrtf(var + 1e-5f);
    h16* op = out + (size_t)orow * C0 + cb;
    for (int i = 0; i < 4; i++) {
        op[i]     = (h16)((a0[i] - mean) * rstd * g[cb + i]     + b[cb + i]);
        op[4 + i] = (h16)((a1[i] - mean) * rstd * g[cb + 4 + i] + b[cb + 4 + i]);
    }
}

// ---------------- generic f16 WMMA GEMM: out[M,N] = A[M,K] @ W[N,K]^T (+epilogue) ----------------
// Double-buffered LDS tiles filled with async global->LDS b128 copies (ASYNCcnt pipeline).
// MODE 0: out f16 = acc + bias
// MODE 1: out f16 = gelu(acc + bias)
// MODE 2: out f32 = acc + bias + resid
template<int MODE>
__global__ void gemm_wmma_kernel(const h16* __restrict__ A, const h16* __restrict__ W,
                                 const float* __restrict__ bias, void* __restrict__ outp,
                                 const float* __restrict__ resid, int M, int N, int K) {
    __shared__ __align__(16) h16 As[2][BM][BK + 8];
    __shared__ __align__(16) h16 Bs[2][BN][BK + 8];
    int tid  = threadIdx.x;
    int lane = tid & 31;
    int wv   = tid >> 5;
    int wm   = wv >> 1;          // 0..3 : 32-row band
    int wn   = wv & 1;           // 0..1 : 64-col band
    int half = lane >> 4;
    int l16  = lane & 15;
    int gm0 = blockIdx.y * BM;
    int gn0 = blockIdx.x * BN;

    // each thread owns 2 chunks of 8 halves (16B) per tile per matrix
    int chunk0 = tid;
    int row0   = chunk0 >> 2;
    int coff0  = (chunk0 & 3) * 8;
    int chunk1 = tid + 256;
    int row1   = chunk1 >> 2;
    int coff1  = (chunk1 & 3) * 8;

    v8f acc[2][4];
    v8f zero = {};
    for (int i = 0; i < 2; i++) for (int j = 0; j < 4; j++) acc[i][j] = zero;

    // prologue: stage tile k0=0 into buffer 0
    ASYNC_CP16(&As[0][row0][coff0], A + (size_t)(gm0 + row0) * K + coff0);
    ASYNC_CP16(&Bs[0][row0][coff0], W + (size_t)(gn0 + row0) * K + coff0);
    ASYNC_CP16(&As[0][row1][coff1], A + (size_t)(gm0 + row1) * K + coff1);
    ASYNC_CP16(&Bs[0][row1][coff1], W + (size_t)(gn0 + row1) * K + coff1);

    int cur = 0;
    for (int k0 = 0; k0 < K; k0 += BK) {
        __syncthreads();            // WAR: everyone done reading buf[cur^1]
        if (k0 + BK < K) {
            int kk = k0 + BK, nb = cur ^ 1;
            ASYNC_CP16(&As[nb][row0][coff0], A + (size_t)(gm0 + row0) * K + kk + coff0);
            ASYNC_CP16(&Bs[nb][row0][coff0], W + (size_t)(gn0 + row0) * K + kk + coff0);
            ASYNC_CP16(&As[nb][row1][coff1], A + (size_t)(gm0 + row1) * K + kk + coff1);
            ASYNC_CP16(&Bs[nb][row1][coff1], W + (size_t)(gn0 + row1) * K + kk + coff1);
            WAIT_ASYNC(4);          // current tile's 4 ops retired; next stage in flight
        } else {
            WAIT_ASYNC(0);
        }
        __syncthreads();            // publish buf[cur] to all waves

        v16h af[2], bf[4];
        for (int mt = 0; mt < 2; mt++) {
            const h16* ar = &As[cur][wm * 32 + mt * 16 + l16][0];
            v8h alo = *(const v8h*)(ar + half * 8);
            v8h ahi = *(const v8h*)(ar + 16 + half * 8);
            for (int i = 0; i < 8; i++) { af[mt][i] = alo[i]; af[mt][8 + i] = ahi[i]; }
        }
        for (int nt = 0; nt < 4; nt++) {
            const h16* br = &Bs[cur][wn * 64 + nt * 16 + l16][0];
            v8h b0 = *(const v8h*)(br + half * 8);
            v8h b1 = *(const v8h*)(br + 16 + half * 8);
            for (int i = 0; i < 8; i++) { bf[nt][i] = b0[i]; bf[nt][8 + i] = b1[i]; }
        }
        for (int mt = 0; mt < 2; mt++)
            for (int nt = 0; nt < 4; nt++)
                acc[mt][nt] = __builtin_amdgcn_wmma_f32_16x16x32_f16(
                    false, af[mt], false, bf[nt], (short)0, acc[mt][nt], false, false);
        cur ^= 1;
    }

    for (int mt = 0; mt < 2; mt++)
        for (int nt = 0; nt < 4; nt++) {
            int gn = gn0 + wn * 64 + nt * 16 + l16;
            float bv = bias[gn];
            for (int r = 0; r < 8; r++) {
                int gm = gm0 + wm * 32 + mt * 16 + r + half * 8;
                size_t oi = (size_t)gm * N + gn;
                float v = acc[mt][nt][r] + bv;
                if (MODE == 0) {
                    ((h16*)outp)[oi] = (h16)v;
                } else if (MODE == 1) {
                    float ge = 0.5f * v * (1.0f + erff(v * 0.70710678118654752f));
                    ((h16*)outp)[oi] = (h16)ge;
                } else {
                    ((float*)outp)[oi] = v + resid[oi];
                }
            }
        }
}

// ---------------- windowed attention: one block per (window, head), 128 threads ----------------
__global__ void attn_kernel(const h16* __restrict__ qkv, const float* __restrict__ rpb,
                            h16* __restrict__ out, int shifted) {
    __shared__ __align__(16) h16 qs[64][40];
    __shared__ __align__(16) h16 ks[64][40];
    __shared__ __align__(16) h16 vT[32][72];
    __shared__ __align__(16) h16 ps[64][72];
    int bid = blockIdx.x;
    int win = bid >> 3;
    int head = bid & 7;
    int tid = threadIdx.x;
    int lane = tid & 31, wv = tid >> 5;
    int half = lane >> 4, l16 = lane & 15;
    const float scale = 0.17677669529663687f;   // 32^-0.5

    for (int i = tid; i < 64 * 32; i += 128) {
        int n = i >> 5, d = i & 31;
        size_t row = (size_t)(win * 64 + n) * 768;
        qs[n][d] = (h16)((float)qkv[row + head * 32 + d] * scale);
        ks[n][d] = qkv[row + 256 + head * 32 + d];
        vT[d][n] = qkv[row + 512 + head * 32 + d];
    }
    __syncthreads();

    // scores: wave owns 16 query rows x 64 key cols (4 WMMA tiles), K = hd = 32
    v8f sc[4];
    v8f zero = {};
    for (int nt = 0; nt < 4; nt++) sc[nt] = zero;
    v16h af;
    {
        const h16* ar = &qs[wv * 16 + l16][0];
        v8h alo = *(const v8h*)(ar + half * 8);
        v8h ahi = *(const v8h*)(ar + 16 + half * 8);
        for (int i = 0; i < 8; i++) { af[i] = alo[i]; af[8 + i] = ahi[i]; }
    }
    for (int nt = 0; nt < 4; nt++) {
        const h16* br = &ks[nt * 16 + l16][0];
        v8h b0 = *(const v8h*)(br + half * 8);
        v8h b1 = *(const v8h*)(br + 16 + half * 8);
        v16h bf;
        for (int i = 0; i < 8; i++) { bf[i] = b0[i]; bf[8 + i] = b1[i]; }
        sc[nt] = __builtin_amdgcn_wmma_f32_16x16x32_f16(false, af, false, bf, (short)0, sc[nt], false, false);
    }

    // relative position bias + shifted-window mask (computed analytically)
    int wh = win >> 6, ww = (win >> 3) & 7, wt = win & 7;
    for (int nt = 0; nt < 4; nt++) {
        int n = nt * 16 + l16;                 // key index in window
        int hn = n >> 4, wn_ = (n >> 2) & 3, tn = n & 3;
        for (int r = 0; r < 8; r++) {
            int m = wv * 16 + r + half * 8;    // query index in window
            int hm = m >> 4, wm_ = (m >> 2) & 3, tm = m & 3;
            int ridx = ((hm - hn + 3) * 7 + (wm_ - wn_ + 3)) * 7 + (tm - tn + 3);
            float v = sc[nt][r] + rpb[ridx * 8 + head];
            if (shifted) {
                int gh0 = (wh << 2) + hm, gw0 = (ww << 2) + wm_, gt0 = (wt << 2) + tm;
                int gh1 = (wh << 2) + hn, gw1 = (ww << 2) + wn_, gt1 = (wt << 2) + tn;
                int la = ((gh0 < 28 ? 0 : (gh0 < 30 ? 1 : 2)) * 3 + (gw0 < 28 ? 0 : (gw0 < 30 ? 1 : 2))) * 3
                         + (gt0 < 28 ? 0 : (gt0 < 30 ? 1 : 2));
                int lb = ((gh1 < 28 ? 0 : (gh1 < 30 ? 1 : 2)) * 3 + (gw1 < 28 ? 0 : (gw1 < 30 ? 1 : 2))) * 3
                         + (gt1 < 28 ? 0 : (gt1 < 30 ? 1 : 2));
                if (la != lb) v -= 100.0f;
            }
            sc[nt][r] = v;
        }
    }

    // row softmax: columns of a row live in lanes sharing the same `half` bit
    for (int r = 0; r < 8; r++) {
        float rmax = -1e30f;
        for (int nt = 0; nt < 4; nt++) rmax = fmaxf(rmax, sc[nt][r]);
        for (int off = 1; off < 16; off <<= 1) rmax = fmaxf(rmax, __shfl_xor(rmax, off, 32));
        float e[4], rsum = 0.f;
        for (int nt = 0; nt < 4; nt++) { e[nt] = __expf(sc[nt][r] - rmax); rsum += e[nt]; }
        for (int off = 1; off < 16; off <<= 1) rsum += __shfl_xor(rsum, off, 32);
        float inv = 1.0f / rsum;
        int m = wv * 16 + r + half * 8;
        for (int nt = 0; nt < 4; nt++) ps[m][nt * 16 + l16] = (h16)(e[nt] * inv);
    }
    __syncthreads();

    // out = P[64,64] @ V[64,32]  (B fragments from vT acting as [N=d][K=m])
    v8f ov[2]; ov[0] = zero; ov[1] = zero;
    for (int k0 = 0; k0 < 64; k0 += 32) {
        v16h pa;
        const h16* ar = &ps[wv * 16 + l16][k0];
        v8h alo = *(const v8h*)(ar + half * 8);
        v8h ahi = *(const v8h*)(ar + 16 + half * 8);
        for (int i = 0; i < 8; i++) { pa[i] = alo[i]; pa[8 + i] = ahi[i]; }
        for (int nt = 0; nt < 2; nt++) {
            const h16* br = &vT[nt * 16 + l16][k0];
            v8h b0 = *(const v8h*)(br + half * 8);
            v8h b1 = *(const v8h*)(br + 16 + half * 8);
            v16h bf;
            for (int i = 0; i < 8; i++) { bf[i] = b0[i]; bf[8 + i] = b1[i]; }
            ov[nt] = __builtin_amdgcn_wmma_f32_16x16x32_f16(false, pa, false, bf, (short)0, ov[nt], false, false);
        }
    }
    for (int nt = 0; nt < 2; nt++)
        for (int r = 0; r < 8; r++) {
            int m = wv * 16 + r + half * 8;
            int d = nt * 16 + l16;
            out[(size_t)(win * 64 + m) * C0 + head * 32 + d] = (h16)ov[nt][r];
        }
}

// ---------------- reverse window partition + un-shift + residual add (fp32) ----------------
__global__ void unpart_res_kernel(const float* __restrict__ xprev, const h16* __restrict__ proj,
                                  float* __restrict__ xout, int shift) {
    int idx = blockIdx.x * 256 + threadIdx.x;       // over L*C
    int tok = idx >> 8, c = idx & 255;
    int h = tok >> 10, w = (tok >> 5) & 31, t = tok & 31;
    int h2 = (h - shift + 32) & 31, w2 = (w - shift + 32) & 31, t2 = (t - shift + 32) & 31;
    int win = ((h2 >> 2) << 6) + ((w2 >> 2) << 3) + (t2 >> 2);
    int n   = ((h2 & 3) << 4) + ((w2 & 3) << 2) + (t2 & 3);
    xout[idx] = xprev[idx] + (float)proj[(size_t)(win * 64 + n) * C0 + c];
}

// ---------------- final LayerNorm + [B,L,C] -> [B,C,H,W,T] transpose ----------------
__global__ void ln_out_kernel(const float* __restrict__ x, const float* __restrict__ g,
                              const float* __restrict__ b, float* __restrict__ out) {
    int wv = threadIdx.x >> 5, lane = threadIdx.x & 31;
    int tok = blockIdx.x * 8 + wv;
    const float* xr = x + (size_t)tok * C0;
    int cb = lane * 8;
    v4f a0 = *(const v4f*)(xr + cb);
    v4f a1 = *(const v4f*)(xr + cb + 4);
    float s = 0.f, ss = 0.f;
    for (int i = 0; i < 4; i++) { s += a0[i] + a1[i]; ss += a0[i]*a0[i] + a1[i]*a1[i]; }
    for (int off = 1; off < 32; off <<= 1) { s += __shfl_xor(s, off, 32); ss += __shfl_xor(ss, off, 32); }
    float mean = s * (1.0f / C0);
    float var  = ss * (1.0f / C0) - mean * mean;
    float rstd = rsqrtf(var + 1e-5f);
    for (int i = 0; i < 8; i++) {
        float v = (i < 4) ? a0[i] : a1[i - 4];
        int c = cb + i;
        out[(size_t)c * LTOK + tok] = (v - mean) * rstd * g[c] + b[c];
    }
}

extern "C" void kernel_launch(void* const* d_in, const int* in_sizes, int n_in,
                              void* d_out, int out_size, void* d_ws, size_t ws_size,
                              hipStream_t stream) {
    (void)in_sizes; (void)n_in; (void)out_size; (void)ws_size;
    const float* x     = (const float*)d_in[0];
    const float* normg = (const float*)d_in[4];
    const float* normb = (const float*)d_in[5];

    const size_t L = LTOK;
    char* wsb = (char*)d_ws;
    size_t off = 0;
    auto alloc = [&](size_t bytes) -> void* {
        void* p = wsb + off;
        off += (bytes + 255) & ~(size_t)255;
        return p;
    };
    h16* wqkv16[2]; h16* wproj16[2]; h16* wfc1_16[2]; h16* wfc2_16[2];
    for (int blk = 0; blk < 2; blk++) {
        wqkv16[blk]  = (h16*)alloc(768 * 256 * sizeof(h16));
        wproj16[blk] = (h16*)alloc(256 * 256 * sizeof(h16));
        wfc1_16[blk] = (h16*)alloc(1024 * 256 * sizeof(h16));
        wfc2_16[blk] = (h16*)alloc(256 * 1024 * sizeof(h16));
    }
    h16*   XW   = (h16*)alloc(L * 256 * sizeof(h16));
    h16*   QKV  = (h16*)alloc(L * 768 * sizeof(h16));
    h16*   ATT  = (h16*)alloc(L * 256 * sizeof(h16));
    h16*   PROJ = (h16*)alloc(L * 256 * sizeof(h16));
    h16*   HLN  = (h16*)alloc(L * 256 * sizeof(h16));
    h16*   FC1  = (h16*)alloc(L * 1024 * sizeof(h16));
    float* XA   = (float*)alloc(L * 256 * sizeof(float));
    float* XB   = (float*)alloc(L * 256 * sizeof(float));

    const float* xprev = x;
    for (int blk = 0; blk < 2; blk++) {
        int base = (blk == 0) ? 6 : 19;
        const float* n1g   = (const float*)d_in[base + 0];
        const float* n1b   = (const float*)d_in[base + 1];
        const float* qkvw  = (const float*)d_in[base + 2];
        const float* qkvb  = (const float*)d_in[base + 3];
        const float* rpb   = (const float*)d_in[base + 4];
        const float* projw = (const float*)d_in[base + 5];
        const float* projb = (const float*)d_in[base + 6];
        const float* n2g   = (const float*)d_in[base + 7];
        const float* n2b   = (const float*)d_in[base + 8];
        const float* fc1w  = (const float*)d_in[base + 9];
        const float* fc1b  = (const float*)d_in[base + 10];
        const float* fc2w  = (const float*)d_in[base + 11];
        const float* fc2b  = (const float*)d_in[base + 12];
        int shift = (blk == 0) ? 0 : 2;

        cvt_f16_kernel<<<(768 * 256 + 255) / 256, 256, 0, stream>>>(qkvw, wqkv16[blk], 768 * 256);
        cvt_f16_kernel<<<(256 * 256 + 255) / 256, 256, 0, stream>>>(projw, wproj16[blk], 256 * 256);
        cvt_f16_kernel<<<(1024 * 256 + 255) / 256, 256, 0, stream>>>(fc1w, wfc1_16[blk], 1024 * 256);
        cvt_f16_kernel<<<(256 * 1024 + 255) / 256, 256, 0, stream>>>(fc2w, wfc2_16[blk], 256 * 1024);

        // LN1 + shift + partition
        ln_part_kernel<<<L / 8, 256, 0, stream>>>(xprev, n1g, n1b, XW, shift, 1);
        // QKV GEMM [L,256] x [768,256]^T
        gemm_wmma_kernel<0><<<dim3(768 / BN, L / BM), 256, 0, stream>>>(
            XW, wqkv16[blk], qkvb, QKV, nullptr, (int)L, 768, 256);
        // windowed attention
        attn_kernel<<<512 * 8, 128, 0, stream>>>(QKV, rpb, ATT, shift ? 1 : 0);
        // proj GEMM [L,256] x [256,256]^T
        gemm_wmma_kernel<0><<<dim3(256 / BN, L / BM), 256, 0, stream>>>(
            ATT, wproj16[blk], projb, PROJ, nullptr, (int)L, 256, 256);
        // reverse partition + un-shift + residual
        unpart_res_kernel<<<(L * 256) / 256, 256, 0, stream>>>(xprev, PROJ, XA, shift);
        // LN2 (plain order)
        ln_part_kernel<<<L / 8, 256, 0, stream>>>(XA, n2g, n2b, HLN, 0, 0);
        // fc1 GEMM + GELU
        gemm_wmma_kernel<1><<<dim3(1024 / BN, L / BM), 256, 0, stream>>>(
            HLN, wfc1_16[blk], fc1b, FC1, nullptr, (int)L, 1024, 256);
        // fc2 GEMM + bias + residual -> fp32 stream
        gemm_wmma_kernel<2><<<dim3(256 / BN, L / BM), 256, 0, stream>>>(
            FC1, wfc2_16[blk], fc2b, XB, XA, (int)L, 256, 1024);
        xprev = XB;
    }
    // final LN + transpose to [B,C,H,W,T]
    ln_out_kernel<<<L / 8, 256, 0, stream>>>(XB, normg, normb, (float*)d_out);
}
